// CPLSTM_1245540516200
// MI455X (gfx1250) — matrix operand order repaired and checked by reading, not compiled, not run on hardware
//
#include <hip/hip_runtime.h>
#include <hip/hip_bf16.h>

// ---------------------------------------------------------------------------
// CP-LSTM for MI455X (gfx1250).
//  Phase A: convert E/a/b/ct to bf16 (a,b,ct transposed so WMMA B-fragments are
//           two contiguous 16B chunks per lane), zero h0 and the grid-barrier counter.
//  Phase B: XB = (E[inp]) @ b  -- 16384x2048, K=1024, bf16 WMMA 16x16x32.
//  Phase C: persistent 16-WG kernel, 512 serial steps:
//           TDM-stage h (64KB) into LDS,    p1: gates slice = (h@a)*XB[t] (WMMA)
//           TDM-stage gates (128KB) to LDS, p2: g = gates_k @ ct4[k]      (WMMA)
//           p3: sigmoid/tanh, c/h update in LDS, write hidden_seq.
//  Shared operands (h, gates) are staged once per WG per step with the Tensor
//  Data Mover instead of being re-read 4-8x from L2 by every wave; weights
//  (aT/ctT, 8MB bf16) stream from the 192MB L2 each step.
// ---------------------------------------------------------------------------

#define DD   1024
#define HH   1024
#define RR   512
#define G4R  2048
#define VV   32000
#define BB   32
#define TT   512
#define NWG  16

typedef __attribute__((ext_vector_type(16))) __bf16 v16bf;
typedef __attribute__((ext_vector_type(8)))  float  v8f;
typedef __attribute__((ext_vector_type(4)))  unsigned v4u;
typedef __attribute__((ext_vector_type(8)))  int      v8i_;
typedef __attribute__((ext_vector_type(4)))  int      v4i_;

union FragBF { v16bf v; uint4 q[2]; };

__device__ __forceinline__ float sigmoidf_(float x) {
  return 1.0f / (1.0f + __expf(-x));
}

// WMMA wrapper: D = A*B + C, bf16 inputs, f32 accumulate (16x16x32).
__device__ __forceinline__ v8f wmma_bf16(v16bf a, v16bf b, v8f c) {
  return __builtin_amdgcn_wmma_f32_16x16x32_bf16(
      false, a, false, b, (short)0, c, false, false);
}

// A fragment (16x32 bf16, ISA 7.12.2): lane = row (lane&15); two contiguous
// 16B chunks at K = hs*8 and K = 16+hs*8.
__device__ __forceinline__ v16bf load_a_frag(const __bf16* rowptr, int hs) {
  FragBF f;
  f.q[0] = *(const uint4*)(rowptr + hs * 8);
  f.q[1] = *(const uint4*)(rowptr + 16 + hs * 8);
  return f.v;
}

// B fragment (32x16 bf16) from a TRANSPOSED [N][K] matrix: lane = column
// (lane&15); two contiguous 16B chunks at K = hs*8 and K = 16+hs*8.
__device__ __forceinline__ v16bf load_b_frag(const __bf16* colptr, int hs) {
  FragBF f;
  f.q[0] = *(const uint4*)(colptr + hs * 8);
  f.q[1] = *(const uint4*)(colptr + 16 + hs * 8);
  return f.v;
}

// ---------------------------------------------------------------------------
// TDM: DMA a 2-D tile (rows x row_qwords u64 elements, contiguous rows) from
// global memory into LDS.  D# packing per CDNA5 ISA:
//   group0: count=1 | lds_addr | global_addr[56:0] | type=2
//   group1: data_size=3 (8B), tensor_dim0/1, tile_dim0/1, dim0_stride
// Issued by one wave; completion via s_wait_tensorcnt.
// This toolchain exposes the 6-arg builtin:
//   (uint32x4 g0, int32x8 g1, int32x4 g2, int32x4 g3, int32x8 unused, i32 cpol)
// ---------------------------------------------------------------------------
#if __has_builtin(__builtin_amdgcn_tensor_load_to_lds) && \
    __has_builtin(__builtin_amdgcn_s_wait_tensorcnt)
#define HAVE_TDM 1
__device__ __forceinline__ void tdm_load_2d(unsigned lds_off, const void* gsrc,
                                            unsigned rows, unsigned row_qw) {
  unsigned long long ga = (unsigned long long)(uintptr_t)gsrc;
  v4u g0;
  g0[0] = 1u;                                           // count=1, user mode
  g0[1] = lds_off;                                      // lds_addr (bytes)
  g0[2] = (unsigned)(ga & 0xFFFFFFFFu);                 // global_addr[31:0]
  g0[3] = (unsigned)((ga >> 32) & 0x01FFFFFFu)          // global_addr[56:32]
          | (2u << 30);                                 // type=2 ("image")
  v8i_ g1;
  g1[0] = (int)(3u << 16);                              // data_size=8B, mask=0
  g1[1] = (int)(row_qw << 16);                          // tensor_dim0 lo16
  g1[2] = (int)((row_qw >> 16) & 0xFFFFu)               // tensor_dim0 hi16
          | (int)(rows << 16);                          // tensor_dim1 lo16
  g1[3] = (int)(row_qw << 16);                          // dim1 hi=0, tile_dim0
  g1[4] = (int)rows;                                    // tile_dim1; tile_dim2=0
  g1[5] = (int)row_qw;                                  // tensor_dim0_stride lo
  g1[6] = 0;                                            // stride hi / dim1_stride
  g1[7] = 0;
  v4i_ z4 = {0, 0, 0, 0};
  v8i_ z8 = {0, 0, 0, 0, 0, 0, 0, 0};
  __builtin_amdgcn_tensor_load_to_lds(g0, g1, z4, z4, z8, 0);
}
#else
#define HAVE_TDM 0
#endif

// Stage a [rows x row_qw u64] tile into LDS.  TDM path: wave 0 issues the DMA
// and waits TENSORcnt; fallback: cooperative b128 copy.  Barrier after either.
__device__ __forceinline__ void stage_to_lds(__bf16* dst_lds,
                                             const __bf16* src, int n_elems,
                                             unsigned rows, unsigned row_qw,
                                             int tid, int nthr) {
#if HAVE_TDM
  (void)n_elems; (void)nthr;
  if ((tid >> 5) == 0) {
    tdm_load_2d((unsigned)(uintptr_t)dst_lds, src, rows, row_qw);
    __builtin_amdgcn_s_wait_tensorcnt((short)0);
  }
#else
  (void)rows; (void)row_qw;
  for (int i = tid * 8; i < n_elems; i += nthr * 8)
    *(uint4*)(dst_lds + i) = *(const uint4*)(src + i);
#endif
  __syncthreads();
}

// ---------------------------------------------------------------------------
// Phase A: conversions / transposes / state init.
// ---------------------------------------------------------------------------
__global__ void cvt_init_kernel(const float* __restrict__ E,
                                const float* __restrict__ a,
                                const float* __restrict__ b,
                                const float* __restrict__ ct,
                                __bf16* __restrict__ E_bf,
                                __bf16* __restrict__ aT,   // [G4R][DD]
                                __bf16* __restrict__ bT,   // [G4R][DD]
                                __bf16* __restrict__ ctT,  // [HH][G4R]
                                __bf16* __restrict__ h_bf, // [BB][HH]
                                unsigned* __restrict__ bar) {
  long i0 = (long)blockIdx.x * blockDim.x + threadIdx.x;
  long stride = (long)gridDim.x * blockDim.x;
  const long nE = (long)VV * DD;
  for (long i = i0; i < nE; i += stride) E_bf[i] = (__bf16)E[i];
  const long nW = (long)G4R * DD;
  for (long i = i0; i < nW; i += stride) {
    long n = i / DD, k = i % DD;       // aT[n][k] = a[k][n]
    aT[i] = (__bf16)a[k * G4R + n];
    bT[i] = (__bf16)b[k * G4R + n];
  }
  const long nC = (long)HH * G4R;
  for (long i = i0; i < nC; i += stride) {
    long h = i / G4R, kr = i % G4R;    // ctT[h][kr] = ct[kr][h]
    ctT[i] = (__bf16)ct[kr * HH + h];
  }
  for (long i = i0; i < (long)BB * HH; i += stride) h_bf[i] = (__bf16)0.0f;
  if (i0 == 0) bar[0] = 0u;            // reset grid barrier each launch
}

// ---------------------------------------------------------------------------
// Phase B: XB[t*B + b][n] = sum_k E_bf[inp[b][t]][k] * bT[n][k]   (bf16 out)
// Block = 32(M) x 64(N) tile, 8 waves, each wave one 16x16 tile, K-loop 32.
// ---------------------------------------------------------------------------
__global__ __launch_bounds__(256) void xb_gemm_kernel(
    const int* __restrict__ inp, const __bf16* __restrict__ E_bf,
    const __bf16* __restrict__ bT, __bf16* __restrict__ XB) {
  const int blk = blockIdx.x;
  const int bm = blk / (G4R / 64);   // 0..511
  const int bn = blk % (G4R / 64);   // 0..31
  const int w = threadIdx.x >> 5;
  const int lane = threadIdx.x & 31;
  const int hs = lane >> 4, ln = lane & 15;
  const int m0 = bm * 32 + (w >> 2) * 16;
  const int n0 = bn * 64 + (w & 3) * 16;

  // Gather: row (t,b) of x comes from E[inp[b][t]].
  const int row = m0 + ln;
  const int t = row >> 5, bidx = row & 31;
  const int token = inp[bidx * TT + t];
  const __bf16* arow = E_bf + (long)token * DD;
  const __bf16* bcol = bT + (long)(n0 + ln) * DD;

  v8f acc = {};
  for (int kb = 0; kb < DD; kb += 32) {
    __builtin_prefetch(bcol + kb + 64, 0, 3);
    v16bf af = load_a_frag(arow + kb, hs);
    v16bf bf = load_b_frag(bcol + kb, hs);
    acc = wmma_bf16(af, bf, acc);
  }
#pragma unroll
  for (int v = 0; v < 8; ++v) {
    int r = m0 + v + 8 * hs;           // C layout: VGPR v -> M = v + 8*hs
    XB[(long)r * G4R + n0 + ln] = (__bf16)acc[v];
  }
}

// ---------------------------------------------------------------------------
// Grid barrier: monotonic counter + acquire fence; also ticks the CDNA5
// cluster barrier (NOP when not dispatched as a cluster).
// ---------------------------------------------------------------------------
__device__ __forceinline__ void grid_barrier(unsigned* bar, unsigned target) {
  __syncthreads();
  if (threadIdx.x == 0) {
    __threadfence();                   // release our stores to device scope
    atomicAdd(bar, 1u);
    while (__hip_atomic_load(bar, __ATOMIC_RELAXED,
                             __HIP_MEMORY_SCOPE_AGENT) < target) {
      __builtin_amdgcn_s_sleep(2);
    }
  }
  __syncthreads();
  __threadfence();                     // acquire: invalidate stale L0 lines
  __builtin_amdgcn_s_cluster_barrier();
}

// ---------------------------------------------------------------------------
// Phase C: persistent recurrent kernel. 16 WGs x 512 threads (16 waves).
// WG wg owns gates columns [wg*128, +128) and h columns [wg*64, +64).
// Dynamic LDS: h (64KB) | gates (128KB) | g (32KB) | c (8KB) = 232KB / WGP.
// ---------------------------------------------------------------------------
__global__ __launch_bounds__(512) void rnn_kernel(
    const __bf16* __restrict__ XB, const __bf16* __restrict__ aT,
    const __bf16* __restrict__ ctT, __bf16* __restrict__ h_bf,
    __bf16* __restrict__ gates_bf, unsigned* __restrict__ bar,
    float* __restrict__ out) {
  extern __shared__ char smem[];
  __bf16* h_lds     = (__bf16*)smem;                               // 64KB
  __bf16* gates_lds = (__bf16*)(smem + 65536);                     // 128KB
  float*  g_lds     = (float*)(smem + 65536 + 131072);             // 32KB
  float*  c_lds     = (float*)(smem + 65536 + 131072 + 32768);     // 8KB

  const int wg = blockIdx.x;
  const int tid = threadIdx.x;
  const int w = tid >> 5, lane = tid & 31;
  const int hs = lane >> 4, ln = lane & 15;

  for (int i = tid; i < 32 * 64; i += 512) c_lds[i] = 0.0f;
  __syncthreads();

  float* seq = out;                            // (B, T, H)
  float* hT = out + (long)BB * TT * HH;        // (B, H)
  float* cT = hT + (long)BB * HH;              // (B, H)

  // Phase-1 tile for this wave: 2(M) x 8(N) grid of 16x16 tiles over 32x128.
  const int p1_m0 = (w & 1) * 16;
  const int p1_n0 = wg * 128 + (w >> 1) * 16;

  unsigned bar_target = 0;

  for (int t = 0; t < TT; ++t) {
    // Stage h (32 x 1024 bf16) into LDS once per WG (TDM DMA).
    stage_to_lds(h_lds, h_bf, BB * HH, BB, HH * 2 / 8, tid, 512);

    // ---- phase 1: gates[:, slice] = (h @ a)[:, slice] * XB[t][:, slice]
    {
      const __bf16* arow0 = h_lds + (p1_m0 + ln) * HH;
      const __bf16* bcol = aT + (long)(p1_n0 + ln) * HH;
      v8f acc = {};
      for (int kb = 0; kb < HH; kb += 32) {
        __builtin_prefetch(bcol + kb + 64, 0, 3);
        v16bf af = load_a_frag(arow0 + kb, hs);   // ds_load_b128 x2
        v16bf bf = load_b_frag(bcol + kb, hs);    // global_load_b128 x2
        acc = wmma_bf16(af, bf, acc);
      }
      const __bf16* xbrow = XB + (long)t * BB * G4R;
#pragma unroll
      for (int v = 0; v < 8; ++v) {
        int m = p1_m0 + v + 8 * hs;
        int col = p1_n0 + ln;
        float g = acc[v] * (float)xbrow[(long)m * G4R + col];
        gates_bf[m * G4R + col] = (__bf16)g;
      }
    }
    bar_target += NWG;
    grid_barrier(bar, bar_target);

    // Stage full gates (32 x 2048 bf16) into LDS once per WG (TDM DMA).
    stage_to_lds(gates_lds, gates_bf, BB * G4R, BB, G4R * 2 / 8, tid, 512);

    // ---- phase 2: g[k][:, h-slice] = gates_k @ ct4[k][:, h-slice]
#pragma unroll
    for (int i = 0; i < 2; ++i) {
      const int tt2 = w * 2 + i;         // 32 tiles: 4 gates x (2M x 4N)
      const int k = tt2 >> 3;
      const int r = tt2 & 7;
      const int m0 = (r >> 2) * 16;
      const int nl0 = (r & 3) * 16;      // local h column base within slice
      const int n0h = wg * 64 + nl0;
      const __bf16* arow0 = gates_lds + (m0 + ln) * G4R + k * RR;
      const __bf16* bcol = ctT + (long)(n0h + ln) * G4R + k * RR;
      v8f acc = {};
      for (int rb = 0; rb < RR; rb += 32) {
        __builtin_prefetch(bcol + rb + 64, 0, 3);
        v16bf af = load_a_frag(arow0 + rb, hs);   // ds_load_b128 x2
        v16bf bf = load_b_frag(bcol + rb, hs);    // global_load_b128 x2
        acc = wmma_bf16(af, bf, acc);
      }
#pragma unroll
      for (int v = 0; v < 8; ++v) {
        int m = m0 + v + 8 * hs;
        g_lds[k * 2048 + m * 64 + nl0 + ln] = acc[v];
      }
    }
    __syncthreads();

    // ---- phase 3: activations + state update for 32x64 slice
#pragma unroll
    for (int j = 0; j < 4; ++j) {
      const int e = tid * 4 + j;         // 2048 elements / 512 threads
      const int m = e >> 6, n = e & 63;
      float gf = sigmoidf_(g_lds[0 * 2048 + m * 64 + n]);
      float gi = sigmoidf_(g_lds[1 * 2048 + m * 64 + n]);
      float gg = tanhf(g_lds[2 * 2048 + m * 64 + n]);
      float go = sigmoidf_(g_lds[3 * 2048 + m * 64 + n]);
      float c = gf * c_lds[m * 64 + n] + gi * gg;
      c_lds[m * 64 + n] = c;
      float h = go * tanhf(c);
      const int col = wg * 64 + n;
      seq[((long)m * TT + t) * HH + col] = h;
      h_bf[m * HH + col] = (__bf16)h;
      if (t == TT - 1) {
        hT[m * HH + col] = h;
        cT[m * HH + col] = c;
      }
    }
    bar_target += NWG;
    grid_barrier(bar, bar_target);       // publish h before next step
  }
}

// ---------------------------------------------------------------------------
extern "C" void kernel_launch(void* const* d_in, const int* in_sizes, int n_in,
                              void* d_out, int out_size, void* d_ws,
                              size_t ws_size, hipStream_t stream) {
  (void)in_sizes; (void)n_in; (void)out_size; (void)ws_size;
  const int*   inp = (const int*)d_in[0];     // (B, T) int32
  const float* E   = (const float*)d_in[1];   // (V, D)
  const float* a   = (const float*)d_in[2];   // (H, 4R)
  const float* b   = (const float*)d_in[3];   // (D, 4R)
  const float* ct  = (const float*)d_in[4];   // (4R, H)

  char* ws = (char*)d_ws;
  size_t off = 0;
  auto alloc = [&](size_t bytes) -> void* {
    void* p = ws + off;
    off += (bytes + 255) & ~(size_t)255;
    return p;
  };
  __bf16* E_bf     = (__bf16*)alloc((size_t)VV * DD * 2);        // 65.5 MB
  __bf16* aT       = (__bf16*)alloc((size_t)G4R * DD * 2);       //  4.2 MB
  __bf16* bT       = (__bf16*)alloc((size_t)G4R * DD * 2);       //  4.2 MB
  __bf16* ctT      = (__bf16*)alloc((size_t)HH * G4R * 2);       //  4.2 MB
  __bf16* XB       = (__bf16*)alloc((size_t)TT * BB * G4R * 2);  // 67.1 MB
  __bf16* h_bf     = (__bf16*)alloc((size_t)BB * HH * 2);
  __bf16* gates_bf = (__bf16*)alloc((size_t)BB * G4R * 2);
  unsigned* bar    = (unsigned*)alloc(256);

  cvt_init_kernel<<<4096, 256, 0, stream>>>(E, a, b, ct, E_bf, aT, bT, ctT,
                                            h_bf, bar);
  xb_gemm_kernel<<<(TT * BB / 32) * (G4R / 64), 256, 0, stream>>>(inp, E_bf,
                                                                  bT, XB);
  const size_t lds_bytes = 65536 + 131072 + 32768 + 8192;  // 232KB / WGP
  rnn_kernel<<<NWG, 512, lds_bytes, stream>>>(XB, aT, ctT, h_bf, gates_bf,
                                              bar, (float*)d_out);
}
